// DynamicNet_33715493274267
// MI455X (gfx1250) — matrix operand order repaired and checked by reading, not compiled, not run on hardware
//
#include <hip/hip_runtime.h>

// ---------------------------------------------------------------------------
// Problem constants (fixed genome topology)
// ---------------------------------------------------------------------------
#define B_BATCH 4096
#define N_IN    512
#define H1      1536
#define H2      1536
#define N_OUT   512
#define NV_COLS (N_IN + H1 + H2)   // 3584
#define FAN     32

typedef __attribute__((ext_vector_type(16))) __bf16 v16bf;
typedef __attribute__((ext_vector_type(8)))  __bf16 v8bf;
typedef __attribute__((ext_vector_type(8)))  float  v8f;
typedef __attribute__((ext_vector_type(4)))  int    v4i;

// Async global->LDS path (CDNA5). Guarded so the file compiles on toolchains
// (and in the host pass) where the builtins are not declared.
#if defined(__AMDGCN__) && \
    __has_builtin(__builtin_amdgcn_global_load_async_to_lds_b128) && \
    __has_builtin(__builtin_amdgcn_s_wait_asynccnt)
#define ASYNC_OK 1
#else
#define ASYNC_OK 0
#endif

// ---------------------------------------------------------------------------
// Small utility kernels
// ---------------------------------------------------------------------------
__global__ void zero_f32_kernel(float* __restrict__ p, int n) {
    int i = blockIdx.x * blockDim.x + threadIdx.x;
    if (i < n) p[i] = 0.0f;
}

__global__ void densify_kernel(const int* __restrict__ src,
                               const int* __restrict__ dst,
                               const float* __restrict__ w,
                               int E, float* __restrict__ Wt, int K) {
    int e = blockIdx.x * blockDim.x + threadIdx.x;
    if (e < E) {
        atomicAdd(&Wt[(size_t)dst[e] * (size_t)K + (size_t)src[e]], w[e]);
    }
}

__global__ void f32_to_bf16_kernel(const float* __restrict__ in,
                                   __bf16* __restrict__ out, int n) {
    int i = blockIdx.x * blockDim.x + threadIdx.x;
    if (i < n) out[i] = (__bf16)in[i];
}

__global__ void pack_x_kernel(const float* __restrict__ x,
                              __bf16* __restrict__ nvb) {
    int i = blockIdx.x * blockDim.x + threadIdx.x;
    if (i < B_BATCH * N_IN) {
        int r = i / N_IN;
        int c = i - r * N_IN;
        nvb[(size_t)r * NV_COLS + c] = (__bf16)x[i];
    }
}

// ---------------------------------------------------------------------------
// BF16 WMMA GEMM with fused ReLU, LDS double-buffered K pipeline.
//   C[M x N] = relu(A[M x K] * W[K x N]),  W given transposed: Wt[N][K].
//
// Block tile 128 x (64*WNI), 256 threads = 8 waves (wave32) arranged 2(M)x4(N);
// each wave computes 64 x (16*WNI) = 4 x WNI WMMA tiles, K stepped by 32.
//   WNI=4 : block 128x256, 16 WMMAs per 16 ds_load_b128 per K-step.
//   WNI=2 : block 128x128 (for the narrow output level).
//
// LDS per buffer: A tile 128x32 bf16 + B tile (64*WNI)x32 bf16, rows padded
// to 40 elems (80B) so the per-lane 16B ds_load_b128 fragment reads are
// bank-conflict free (20*r mod 64 covers all multiples of 4).
//
// Fragment layouts per CDNA5 ISA (wave32):
//   A 16x32 bf16 : lane L (m = L&15): if L<16 holds K {0..7, 16..23},
//                  else K {8..15, 24..31}.
//   B 32x16 bf16 : lane L (n = L&15): if L<16 holds K {0..7, 16..23}, else K {8..15, 24..31}.
//   C/D 16x16 f32: VGPR r, lane L: row = r + 8*(L>=16), col = L&15.
// ---------------------------------------------------------------------------
#define M_TILE 128
#define KSTEP  32
#define A_LD   40                         // bf16 elems per LDS row (80B)
#define B_LD   40

template <int WNI>
__global__ __launch_bounds__(256)
void gemm_relu_bf16_kernel(const __bf16* __restrict__ A, int lda, int K,
                           const __bf16* __restrict__ Wt,          // [N][K]
                           __bf16* __restrict__ outB, int ldob,    // bf16 out (or null)
                           float*  __restrict__ outF, int ldof) {  // f32 out (or null)
    constexpr int NT     = 64 * WNI;                  // block N tile
    constexpr int TILE_E = M_TILE * A_LD + NT * B_LD; // elems per LDS buffer
    constexpr int BOPS   = NT / 128;                  // B row-groups per thread
    constexpr int OPS    = 2 + 2 * BOPS;              // async ops / thread / stage

    __shared__ __bf16 lds[2 * TILE_E];

    const int t       = threadIdx.x;
    const int lane    = t & 31;
    const int wave    = t >> 5;
    const int halfsel = lane >> 4;
    const int l15     = lane & 15;

    const int m0 = blockIdx.x * M_TILE;
    const int n0 = blockIdx.y * NT;

    // --- staging assignment: thread t covers 32B (16 elems) of one A row and
    // of BOPS rows of B.
    const int sRow  = t >> 1;            // 0..127
    const int sHalf = (t & 1) * 16;      // elem offset 0 or 16 within K-step
    const __bf16* gA = A + (size_t)(m0 + sRow) * (size_t)lda + sHalf;
    __bf16* lA0 = lds + sRow * A_LD + sHalf;
    const __bf16* gBv[BOPS];
    __bf16*       lBv[BOPS];
#pragma unroll
    for (int it = 0; it < BOPS; ++it) {
        int r = sRow + 128 * it;
        gBv[it] = Wt + (size_t)(n0 + r) * (size_t)K + sHalf;
        lBv[it] = lds + M_TILE * A_LD + r * B_LD + sHalf;
    }

    // --- per-wave fragment origins in LDS
    const int wmBase = (wave >> 2) * 64;          // wave M origin within tile
    const int wnBase = (wave & 3) * (16 * WNI);   // wave N origin within tile

    v8f acc[4][WNI] = {};

    const int S = K / KSTEP;

    // Compute one K-step from LDS buffer `buf`.
    auto compute = [&](int buf) {
        const __bf16* aB = lds + buf * TILE_E
                         + (wmBase + l15) * A_LD + halfsel * 8;
        const __bf16* bB = lds + buf * TILE_E + M_TILE * A_LD
                         + (wnBase + l15) * B_LD + halfsel * 8;
        v16bf afrag[4];
        v16bf bfrag[WNI];
#pragma unroll
        for (int i = 0; i < 4; ++i) {
            const __bf16* p = aB + i * 16 * A_LD;
            union { v16bf v; v8bf h[2]; } u;
            u.h[0] = *(const v8bf*)(p);
            u.h[1] = *(const v8bf*)(p + 16);
            afrag[i] = u.v;
        }
#pragma unroll
        for (int j = 0; j < WNI; ++j) {
            const __bf16* p = bB + j * 16 * B_LD;
            union { v16bf v; v8bf h[2]; } u;
            u.h[0] = *(const v8bf*)(p);
            u.h[1] = *(const v8bf*)(p + 16);
            bfrag[j] = u.v;
        }
#pragma unroll
        for (int i = 0; i < 4; ++i) {
#pragma unroll
            for (int j = 0; j < WNI; ++j) {
                acc[i][j] = __builtin_amdgcn_wmma_f32_16x16x32_bf16(
                    false, afrag[i], false, bfrag[j],
                    (short)0, acc[i][j], false, false);
            }
        }
    };

#if ASYNC_OK
    // -------- async global->LDS double-buffered pipeline --------
    auto issue_stage = [&](int buf, int k) {
        const __bf16* ga = gA + k;
        __bf16* la = lA0 + buf * TILE_E;
        __builtin_amdgcn_global_load_async_to_lds_b128(
            (v4i*)(ga),     (v4i*)(la),     0, 0);
        __builtin_amdgcn_global_load_async_to_lds_b128(
            (v4i*)(ga + 8), (v4i*)(la + 8), 0, 0);
#pragma unroll
        for (int it = 0; it < BOPS; ++it) {
            const __bf16* gb = gBv[it] + k;
            __bf16* lb = lBv[it] + buf * TILE_E;
            __builtin_amdgcn_global_load_async_to_lds_b128(
                (v4i*)(gb),     (v4i*)(lb),     0, 0);
            __builtin_amdgcn_global_load_async_to_lds_b128(
                (v4i*)(gb + 8), (v4i*)(lb + 8), 0, 0);
        }
    };

    issue_stage(0, 0);
#pragma unroll 2
    for (int s = 0; s < S; ++s) {
        const int buf = s & 1;
        if (s + 1 < S) {
            issue_stage(buf ^ 1, (s + 1) * KSTEP);
            __builtin_amdgcn_s_wait_asynccnt(OPS);   // stage s complete
        } else {
            __builtin_amdgcn_s_wait_asynccnt(0);
        }
        __syncthreads();
        compute(buf);
        __syncthreads();
    }
#else
    // -------- fallback: register prefetch + ds_store staging --------
    v8bf ra0, ra1, rb0[BOPS], rb1[BOPS];
    auto gload = [&](int k) {
        ra0 = *(const v8bf*)(gA + k);
        ra1 = *(const v8bf*)(gA + k + 8);
#pragma unroll
        for (int it = 0; it < BOPS; ++it) {
            rb0[it] = *(const v8bf*)(gBv[it] + k);
            rb1[it] = *(const v8bf*)(gBv[it] + k + 8);
        }
    };
    auto lstore = [&](int buf) {
        __bf16* la = lA0 + buf * TILE_E;
        *(v8bf*)(la)     = ra0;
        *(v8bf*)(la + 8) = ra1;
#pragma unroll
        for (int it = 0; it < BOPS; ++it) {
            __bf16* lb = lBv[it] + buf * TILE_E;
            *(v8bf*)(lb)     = rb0[it];
            *(v8bf*)(lb + 8) = rb1[it];
        }
    };
    gload(0);
#pragma unroll 2
    for (int s = 0; s < S; ++s) {
        const int buf = s & 1;
        lstore(buf);
        __syncthreads();
        if (s + 1 < S) gload((s + 1) * KSTEP);
        compute(buf);
        __syncthreads();
    }
#endif

    // -------- epilogue: ReLU + store --------
#pragma unroll
    for (int i = 0; i < 4; ++i) {
#pragma unroll
        for (int j = 0; j < WNI; ++j) {
            union { v8f v; float f[8]; } u;
            u.v = acc[i][j];
#pragma unroll
            for (int r = 0; r < 8; ++r) {
                float v = u.f[r];
                v = v > 0.0f ? v : 0.0f;
                int row = m0 + wmBase + 16 * i + r + 8 * halfsel;
                int col = n0 + wnBase + 16 * j + l15;
                if (outF) {
                    outF[(size_t)row * (size_t)ldof + col] = v;
                } else {
                    outB[(size_t)row * (size_t)ldob + col] = (__bf16)v;
                }
            }
        }
    }
}

// ---------------------------------------------------------------------------
// Launch
// ---------------------------------------------------------------------------
extern "C" void kernel_launch(void* const* d_in, const int* in_sizes, int n_in,
                              void* d_out, int out_size, void* d_ws, size_t ws_size,
                              hipStream_t stream) {
    (void)in_sizes; (void)n_in; (void)out_size; (void)ws_size;

    const float* x    = (const float*)d_in[0];
    const int*   src0 = (const int*)  d_in[1];
    const int*   dst0 = (const int*)  d_in[2];
    const float* w0   = (const float*)d_in[3];
    const int*   src1 = (const int*)  d_in[4];
    const int*   dst1 = (const int*)  d_in[5];
    const float* w1   = (const float*)d_in[6];
    const int*   src2 = (const int*)  d_in[7];
    const int*   dst2 = (const int*)  d_in[8];
    const float* w2   = (const float*)d_in[9];
    float* out = (float*)d_out;

    const int E0 = H1 * FAN;
    const int E1 = H2 * FAN;
    const int E2 = N_OUT * FAN;
    const int K0 = N_IN;               // 512
    const int K1 = N_IN + H1;          // 2048
    const int K2 = N_IN + H1 + H2;     // 3584
    const size_t nW0 = (size_t)H1    * K0;
    const size_t nW1 = (size_t)H2    * K1;
    const size_t nW2 = (size_t)N_OUT * K2;

    char* ws = (char*)d_ws;
    size_t off = 0;
    auto carve = [&](size_t bytes) -> void* {
        void* p = ws + off;
        off = (off + bytes + 255) & ~(size_t)255;
        return p;
    };
    __bf16* nvb = (__bf16*)carve((size_t)B_BATCH * NV_COLS * 2);
    float*  W0f = (float*) carve(nW0 * 4);
    float*  W1f = (float*) carve(nW1 * 4);
    float*  W2f = (float*) carve(nW2 * 4);
    __bf16* W0b = (__bf16*)carve(nW0 * 2);
    __bf16* W1b = (__bf16*)carve(nW1 * 2);
    __bf16* W2b = (__bf16*)carve(nW2 * 2);

    const int T = 256;
    zero_f32_kernel<<<(int)((nW0 + T - 1) / T), T, 0, stream>>>(W0f, (int)nW0);
    zero_f32_kernel<<<(int)((nW1 + T - 1) / T), T, 0, stream>>>(W1f, (int)nW1);
    zero_f32_kernel<<<(int)((nW2 + T - 1) / T), T, 0, stream>>>(W2f, (int)nW2);

    densify_kernel<<<(E0 + T - 1) / T, T, 0, stream>>>(src0, dst0, w0, E0, W0f, K0);
    densify_kernel<<<(E1 + T - 1) / T, T, 0, stream>>>(src1, dst1, w1, E1, W1f, K1);
    densify_kernel<<<(E2 + T - 1) / T, T, 0, stream>>>(src2, dst2, w2, E2, W2f, K2);

    f32_to_bf16_kernel<<<(int)((nW0 + T - 1) / T), T, 0, stream>>>(W0f, W0b, (int)nW0);
    f32_to_bf16_kernel<<<(int)((nW1 + T - 1) / T), T, 0, stream>>>(W1f, W1b, (int)nW1);
    f32_to_bf16_kernel<<<(int)((nW2 + T - 1) / T), T, 0, stream>>>(W2f, W2b, (int)nW2);

    pack_x_kernel<<<(B_BATCH * N_IN + T - 1) / T, T, 0, stream>>>(x, nvb);

    {   // Level 1: nv[:, :512] @ W0 -> nv cols [512, 2048)   (block 128x256)
        dim3 grid(B_BATCH / 128, H1 / 256);
        gemm_relu_bf16_kernel<4><<<grid, 256, 0, stream>>>(
            nvb, NV_COLS, K0, W0b, nvb + N_IN, NV_COLS, nullptr, 0);
    }
    {   // Level 2: nv[:, :2048] @ W1 -> nv cols [2048, 3584) (block 128x256)
        dim3 grid(B_BATCH / 128, H2 / 256);
        gemm_relu_bf16_kernel<4><<<grid, 256, 0, stream>>>(
            nvb, NV_COLS, K1, W1b, nvb + (N_IN + H1), NV_COLS, nullptr, 0);
    }
    {   // Output: nv[:, :3584] @ W2 -> d_out fp32            (block 128x128)
        dim3 grid(B_BATCH / 128, N_OUT / 128);
        gemm_relu_bf16_kernel<2><<<grid, 256, 0, stream>>>(
            nvb, NV_COLS, K2, W2b, nullptr, 0, out, N_OUT);
    }
}
